// TransLayer_85899345920257
// MI455X (gfx1250) — matrix operand order repaired and checked by reading, not compiled, not run on hardware
//
#include <hip/hip_runtime.h>
#include <cstdint>

typedef __attribute__((ext_vector_type(16))) _Float16 v16h;
typedef __attribute__((ext_vector_type(8)))  _Float16 v8h;
typedef __attribute__((ext_vector_type(8)))  float    v8f;

#define LDSP 40   // LDS row pitch in f16 (32 data + 8 pad, keeps 16B alignment)

// ---------------- elementwise casts ----------------
__global__ void cast16_kernel(const float* __restrict__ s, _Float16* __restrict__ d, long n){
  long i = (long)blockIdx.x * blockDim.x + threadIdx.x;
  long st = (long)gridDim.x * blockDim.x;
  for (; i < n; i += st) d[i] = (_Float16)s[i];
}

__global__ void cast16_slice_kernel(const float* __restrict__ s, _Float16* __restrict__ d,
                                    long rows, int rowlen, int pitch){
  long n = rows * (long)rowlen;
  long i = (long)blockIdx.x * blockDim.x + threadIdx.x;
  long st = (long)gridDim.x * blockDim.x;
  for (; i < n; i += st){
    long r = i / rowlen; int c = (int)(i - r * rowlen);
    d[r * (long)pitch + c] = (_Float16)s[i];
  }
}

__global__ void cast16_transpose_kernel(const float* __restrict__ s, _Float16* __restrict__ d,
                                        int R, int C){
  int i = blockIdx.x * blockDim.x + threadIdx.x;
  if (i < R * C){ int r = i / C, c = i - r * C; d[(long)c * R + r] = (_Float16)s[i]; }
}

// ---------------- shared WMMA helpers ----------------
// A fragment (16x32 f16): lane L<16 -> row L, K in {hi*8..hi*8+7} U {16+hi*8..}
// B fragment (32x16 f16): lane L<16 -> col L, K in {hi*8..hi*8+7} U {16+hi*8..}
template<int TM, int TN>
__device__ __forceinline__ void mma_tiles(const _Float16* As, const _Float16* Bs,
                                          int arow0, int brow0, int lane, v8f acc[TM][TN])
{
  const int r16 = lane & 15, hi = lane >> 4;
  v16h a[TM], b[TN];
  #pragma unroll
  for (int tm = 0; tm < TM; ++tm){
    const int ra = arow0 + tm * 16 + r16;
    v8h lo = *reinterpret_cast<const v8h*>(As + ra * LDSP + hi * 8);
    v8h h8 = *reinterpret_cast<const v8h*>(As + ra * LDSP + 16 + hi * 8);
    #pragma unroll
    for (int i = 0; i < 8; ++i){ a[tm][i] = lo[i]; a[tm][8 + i] = h8[i]; }
  }
  #pragma unroll
  for (int tn = 0; tn < TN; ++tn){
    const int rb = brow0 + tn * 16 + r16;
    v8h lo = *reinterpret_cast<const v8h*>(Bs + rb * LDSP + hi * 8);
    v8h h8 = *reinterpret_cast<const v8h*>(Bs + rb * LDSP + 16 + hi * 8);
    #pragma unroll
    for (int i = 0; i < 8; ++i){ b[tn][i] = lo[i]; b[tn][8 + i] = h8[i]; }
  }
  #pragma unroll
  for (int tm = 0; tm < TM; ++tm)
    #pragma unroll
    for (int tn = 0; tn < TN; ++tn)
      acc[tm][tn] = __builtin_amdgcn_wmma_f32_16x16x32_f16(
          false, a[tm], false, b[tn], (short)0, acc[tm][tn], false, false);
}

// C/D layout: VGPR r, lanes 0-15: M=r, N=lane; lanes 16-31: M=r+8, N=lane-16
template<int TM, int TN>
__device__ __forceinline__ void store_acc(v8f acc[TM][TN], float* __restrict__ C32,
                                          _Float16* __restrict__ C16,
                                          const float* __restrict__ bp,
                                          int mbase, int nbase, int lane,
                                          int M, int N, int ldc, int relu)
{
  const int r16 = lane & 15, hi = lane >> 4;
  #pragma unroll
  for (int tm = 0; tm < TM; ++tm){
    #pragma unroll
    for (int tn = 0; tn < TN; ++tn){
      const int mb = mbase + tm * 16;
      const int n  = nbase + tn * 16 + r16;
      const float bvv = (bp && n < N) ? bp[n] : 0.f;
      #pragma unroll
      for (int r = 0; r < 8; ++r){
        const int m = mb + r + hi * 8;
        if (m < M && n < N){
          float v = acc[tm][tn][r] + bvv;
          if (relu) v = fmaxf(v, 0.f);
          if (C32) C32[(long)m * ldc + n] = v;
          if (C16) C16[(long)m * ldc + n] = (_Float16)v;
        }
      }
    }
  }
}

// ---------------- batched NT GEMM: C = A[M,K] * W[N,K]^T (+bias)(+relu) ----------------
// Block tile 64x128, 4 waves, 2x4 WMMA tiles per wave. K must be a multiple of 32.
// Out-of-range rows are loaded at a clamped valid address and zero-selected (branch-free).
__global__ void __launch_bounds__(128)
gemm_nt_kernel(const _Float16* __restrict__ A, const _Float16* __restrict__ W,
               const float* __restrict__ bias,
               float* __restrict__ C32, _Float16* __restrict__ C16,
               int M, int N, int K, int lda, int ldw, int ldc,
               long sA, long sW, long sC, long sBias, int wMod, int relu)
{
  __shared__ _Float16 As[64 * LDSP];
  __shared__ _Float16 Bs[128 * LDSP];
  const int t = threadIdx.x, lane = t & 31, wv = t >> 5;
  const int wrow = wv >> 1, wcol = wv & 1;
  const int m0 = blockIdx.y * 64, n0 = blockIdx.x * 128;
  const int batch = blockIdx.z;
  A += (long)batch * sA;
  W += (long)(batch % wMod) * sW;
  if (C32) C32 += (long)batch * sC;
  if (C16) C16 += (long)batch * sC;
  const float* bp = bias ? bias + (long)(batch % wMod) * sBias : nullptr;

  v8f acc[2][4];
  { v8f z = {0.f,0.f,0.f,0.f,0.f,0.f,0.f,0.f};
    #pragma unroll
    for (int i = 0; i < 2; ++i)
      #pragma unroll
      for (int j = 0; j < 4; ++j) acc[i][j] = z; }

  const float4 z4 = {0.f,0.f,0.f,0.f};
  for (int k0 = 0; k0 < K; k0 += 32){
    __syncthreads();
    // A tile: 64x32 f16 = 256 16B chunks, 2 per thread (branch-free clamp+select)
    #pragma unroll
    for (int i = 0; i < 2; ++i){
      const int e = t + i * 128;
      const int row = e >> 2;
      const int cg  = (e & 3) * 8;
      int gm = m0 + row;
      const bool oob = (gm >= M);
      if (oob) gm = M - 1;
      float4 av = *reinterpret_cast<const float4*>(A + (long)gm * lda + k0 + cg);
      if (oob) av = z4;
      *reinterpret_cast<float4*>(&As[row * LDSP + cg]) = av;
      if (k0 + 32 < K)
        __builtin_prefetch(A + (long)gm * lda + k0 + 32 + cg, 0, 1);
    }
    // B tile: 128x32 f16 = 512 16B chunks, 4 per thread
    #pragma unroll
    for (int i = 0; i < 4; ++i){
      const int e = t + i * 128;
      const int row = e >> 2;
      const int cg  = (e & 3) * 8;
      int gn = n0 + row;
      const bool oob = (gn >= N);
      if (oob) gn = N - 1;
      float4 bv = *reinterpret_cast<const float4*>(W + (long)gn * ldw + k0 + cg);
      if (oob) bv = z4;
      *reinterpret_cast<float4*>(&Bs[row * LDSP + cg]) = bv;
    }
    __syncthreads();
    mma_tiles<2,4>(As, Bs, wrow * 32, wcol * 64, lane, acc);
  }
  store_acc<2,4>(acc, C32, C16, bp, m0 + wrow * 32, n0 + wcol * 64, lane, M, N, ldc, relu);
}

// ---------------- batched strided GEMM: C[m,n] = sum_k A(m,k)*B(k,n) ----------------
// 64x64 tile, arbitrary element strides; K guarded (zero-padded in LDS). GCN hops.
__global__ void __launch_bounds__(128)
gemm_str_kernel(const _Float16* __restrict__ A, const _Float16* __restrict__ B,
                float* __restrict__ C32, _Float16* __restrict__ C16,
                int M, int N, int K,
                long sAm, long sAk, long sBk, long sBn, int ldc,
                long strA, long strB, long strC)
{
  __shared__ _Float16 As[64 * LDSP];
  __shared__ _Float16 Bs[64 * LDSP];
  const int t = threadIdx.x, lane = t & 31, wv = t >> 5;
  const int wrow = wv >> 1, wcol = wv & 1;
  const int m0 = blockIdx.y * 64, n0 = blockIdx.x * 64;
  const int batch = blockIdx.z;
  A += (long)batch * strA;
  B += (long)batch * strB;
  if (C32) C32 += (long)batch * strC;
  if (C16) C16 += (long)batch * strC;

  v8f acc[2][2];
  { v8f z = {0.f,0.f,0.f,0.f,0.f,0.f,0.f,0.f};
    acc[0][0] = z; acc[0][1] = z; acc[1][0] = z; acc[1][1] = z; }
  const _Float16 hz = (_Float16)0.f;

  for (int k0 = 0; k0 < K; k0 += 32){
    __syncthreads();
    for (int e = t; e < 64 * 32; e += 128){
      const int row = e >> 5, kk = e & 31;
      int gk = k0 + kk;
      const bool kob = (gk >= K);
      if (kob) gk = K - 1;
      int gm = m0 + row;
      const bool mob = (gm >= M);
      if (mob) gm = M - 1;
      _Float16 avv = A[(long)gm * sAm + (long)gk * sAk];
      if (kob | mob) avv = hz;
      As[row * LDSP + kk] = avv;
      int gn = n0 + row;
      const bool nob = (gn >= N);
      if (nob) gn = N - 1;
      _Float16 bvv = B[(long)gk * sBk + (long)gn * sBn];
      if (kob | nob) bvv = hz;
      Bs[row * LDSP + kk] = bvv;
    }
    __syncthreads();
    mma_tiles<2,2>(As, Bs, wrow * 32, wcol * 32, lane, acc);
  }
  store_acc<2,2>(acc, C32, C16, nullptr, m0 + wrow * 32, n0 + wcol * 32, lane, M, N, ldc, 0);
}

// ---------------- attention: one thread per (b,n,h,l) query row ----------------
__global__ void attn_kernel(const _Float16* __restrict__ Q, const _Float16* __restrict__ Km,
                            const _Float16* __restrict__ Vm, _Float16* __restrict__ O,
                            int Lq, int Lk, int Nn, int Hh, long total)
{
  long tid = (long)blockIdx.x * blockDim.x + threadIdx.x;
  if (tid >= total) return;
  const int l = (int)(tid % Lq);
  long r = tid / Lq;
  const int h = (int)(r % Hh); r /= Hh;
  const int n = (int)(r % Nn); r /= Nn;
  const int b = (int)r;
  const int D = 512, E = 64;
  const long rowQ = ((long)(b * Lq + l) * Nn + n) * D + h * E;
  const long kv0  = ((long)(b * Lk) * Nn + n) * D + h * E;
  const long step = (long)Nn * D;
  const _Float16* qp = Q + rowQ;

  float sc[24];
  #pragma unroll
  for (int s = 0; s < 24; ++s){
    float d = 0.f;
    if (s < Lk){
      const _Float16* kp = Km + kv0 + (long)s * step;
      for (int e = 0; e < E; ++e) d += (float)qp[e] * (float)kp[e];
    }
    sc[s] = d * 0.125f;   // 1/sqrt(64)
  }
  float mx = -1e30f;
  #pragma unroll
  for (int s = 0; s < 24; ++s) if (s < Lk) mx = fmaxf(mx, sc[s]);
  float sum = 0.f;
  #pragma unroll
  for (int s = 0; s < 24; ++s) if (s < Lk){ sc[s] = __expf(sc[s] - mx); sum += sc[s]; }
  const float inv = 1.f / sum;
  #pragma unroll
  for (int s = 0; s < 24; ++s) sc[s] *= inv;

  _Float16* op = O + rowQ;
  for (int e = 0; e < E; ++e){
    float acc = 0.f;
    const _Float16* vp = Vm + kv0 + e;
    #pragma unroll
    for (int s = 0; s < 24; ++s) if (s < Lk) acc += sc[s] * (float)vp[(long)s * step];
    op[e] = (_Float16)acc;
  }
}

// ---------------- residual + LayerNorm (D=512), 128 threads per row ----------------
__device__ __forceinline__ float wave_sum(float v){
  #pragma unroll
  for (int o = 16; o > 0; o >>= 1) v += __shfl_down(v, o, 32);
  return v;
}

__global__ void __launch_bounds__(128)
ln_kernel(const float* __restrict__ a, const float* __restrict__ r,
          const float* __restrict__ g, const float* __restrict__ be,
          float* __restrict__ o32, _Float16* __restrict__ o16)
{
  const int t = threadIdx.x;
  const long base = (long)blockIdx.x * 512;
  float v[4];
  #pragma unroll
  for (int i = 0; i < 4; ++i){
    const int c = t + i * 128;
    float x = a[base + c];
    if (r) x += r[base + c];
    v[i] = x;
  }
  float s = v[0] + v[1] + v[2] + v[3];
  float q = v[0]*v[0] + v[1]*v[1] + v[2]*v[2] + v[3]*v[3];
  __shared__ float red[8];
  const float ws1 = wave_sum(s), ws2 = wave_sum(q);
  if ((t & 31) == 0){ red[t >> 5] = ws1; red[4 + (t >> 5)] = ws2; }
  __syncthreads();
  const float S  = red[0] + red[1] + red[2] + red[3];
  const float Qs = red[4] + red[5] + red[6] + red[7];
  const float mu = S * (1.f / 512.f);
  const float var = Qs * (1.f / 512.f) - mu * mu;
  const float rstd = rsqrtf(var + 1e-5f);
  #pragma unroll
  for (int i = 0; i < 4; ++i){
    const int c = t + i * 128;
    const float y = (v[i] - mu) * rstd * g[c] + be[c];
    if (o32) o32[base + c] = y;
    if (o16) o16[base + c] = (_Float16)y;
  }
}

// ---------------- host orchestration ----------------
extern "C" void kernel_launch(void* const* d_in, const int* in_sizes, int n_in,
                              void* d_out, int out_size, void* d_ws, size_t ws_size,
                              hipStream_t stream)
{
  (void)in_sizes; (void)n_in; (void)out_size; (void)ws_size;
  const int Bc = 16, Lc = 12, LMc = 24, Nc = 207, Dc = 512, Hc = 8, DFFc = 2048;
  const long Ms = (long)Bc * Lc * Nc;     // 39744 rows (b,l,n)
  const long Mm = (long)Bc * LMc * Nc;    // 79488 rows (b,lm,n)
  const int  NBt = Bc * Lc;               // 192 (b,l) batches
  const int  C5  = 5 * Dc;                // 2560

  const float* x        = (const float*)d_in[0];
  const float* memory_  = (const float*)d_in[1];
  const float* data     = (const float*)d_in[2];
  const float* support0 = (const float*)d_in[4];
  const float* wA[8] = {(const float*)d_in[5], (const float*)d_in[6], (const float*)d_in[7],
                        (const float*)d_in[8], (const float*)d_in[9], (const float*)d_in[10],
                        (const float*)d_in[11], (const float*)d_in[12]};
  const float* bA[8] = {(const float*)d_in[13], (const float*)d_in[14], (const float*)d_in[15],
                        (const float*)d_in[16], (const float*)d_in[17], (const float*)d_in[18],
                        (const float*)d_in[19], (const float*)d_in[20]};
  const float* W1 = (const float*)d_in[21]; const float* b1 = (const float*)d_in[22];
  const float* W2 = (const float*)d_in[23]; const float* b2 = (const float*)d_in[24];
  const float* W3 = (const float*)d_in[25]; const float* b3 = (const float*)d_in[26];
  const float* g1 = (const float*)d_in[27]; const float* be1 = (const float*)d_in[28];
  const float* g2 = (const float*)d_in[29]; const float* be2 = (const float*)d_in[30];
  const float* g3 = (const float*)d_in[31]; const float* be3 = (const float*)d_in[32];
  const float* Wg = (const float*)d_in[33]; const float* bg = (const float*)d_in[34];

  float* out    = (float*)d_out;
  float* out1   = out;
  float* outMem = out + (long)Bc * Lc * Nc * Dc;
  float* outDts = outMem + (long)Bc * LMc * Nc * Dc;

  // workspace bump allocator
  char* wsB = (char*)d_ws; size_t off = 0;
  auto alloc  = [&](size_t bytes)->void*{ void* p = wsB + off; off += (bytes + 255) & ~((size_t)255); return p; };
  auto allocH = [&](long n)->_Float16*{ return (_Float16*)alloc((size_t)n * sizeof(_Float16)); };
  auto allocF = [&](long n)->float*   { return (float*)alloc((size_t)n * sizeof(float)); };

  _Float16* xh  = allocH(Ms * Dc);
  _Float16* mh  = allocH(Mm * Dc);
  _Float16* wAh[8]; for (int i = 0; i < 8; ++i) wAh[i] = allocH((long)Dc * Dc);
  _Float16* W1h = allocH((long)DFFc * Dc);
  _Float16* W2h = allocH((long)Dc * DFFc);
  _Float16* W3h = allocH((long)Nc * Dc);
  _Float16* Wgh = allocH((long)Lc * Dc * C5);
  _Float16* StT = allocH((long)Nc * Nc);
  _Float16* qs  = allocH(Ms * Dc);
  _Float16* ks  = allocH(Ms * Dc);
  _Float16* vs  = allocH(Ms * Dc);
  _Float16* at1 = allocH(Ms * Dc);
  float*    o32 = allocF(Ms * Dc);
  float*    x1f = allocF(Ms * Dc);
  _Float16* x1h = allocH(Ms * Dc);
  _Float16* kc  = allocH(Mm * Dc);
  _Float16* vc  = allocH(Mm * Dc);
  float*    x2f = allocF(Ms * Dc);
  _Float16* x2h = allocH(Ms * Dc);
  _Float16* f1h = allocH(Ms * DFFc);
  float*    y32 = allocF(Ms * Dc);
  _Float16* yh  = allocH(Ms * Dc);
  float*    As32 = allocF((long)NBt * Nc * Nc);
  _Float16* Ash  = allocH((long)NBt * Nc * Nc);
  _Float16* hT   = allocH((long)NBt * Nc * C5);
  const long strideT = (long)Nc * C5;

  auto castN = [&](const float* s, _Float16* d, long n){
    long bl = (n + 255) / 256; if (bl > 16384) bl = 16384;
    cast16_kernel<<<(int)bl, 256, 0, stream>>>(s, d, n);
  };
  auto gemmNT = [&](const _Float16* A, const _Float16* Wp, const float* bias,
                    float* C32, _Float16* C16, int M, int N, int K,
                    int lda, int ldw, int ldc,
                    long sA, long sW, long sC, long sBias, int wMod, int relu, int batches){
    dim3 g((unsigned)((N + 127) / 128), (unsigned)((M + 63) / 64), (unsigned)batches);
    gemm_nt_kernel<<<g, 128, 0, stream>>>(A, Wp, bias, C32, C16, M, N, K,
                                          lda, ldw, ldc, sA, sW, sC, sBias, wMod, relu);
  };
  auto gemmSTR = [&](const _Float16* A, const _Float16* B, _Float16* C16,
                     int M, int N, int K, long sAm, long sAk, long sBk, long sBn, int ldc,
                     long strA, long strB, long strC, int batches){
    dim3 g((unsigned)((N + 63) / 64), (unsigned)((M + 63) / 64), (unsigned)batches);
    gemm_str_kernel<<<g, 128, 0, stream>>>(A, B, nullptr, C16, M, N, K,
                                           sAm, sAk, sBk, sBn, ldc, strA, strB, strC);
  };

  // f32 -> f16 staging
  castN(x, xh, Ms * Dc);
  castN(memory_, mh, Mm * Dc);
  for (int i = 0; i < 8; ++i) castN(wA[i], wAh[i], (long)Dc * Dc);
  castN(W1, W1h, (long)DFFc * Dc);
  castN(W2, W2h, (long)Dc * DFFc);
  castN(W3, W3h, (long)Nc * Dc);
  castN(Wg, Wgh, (long)Lc * Dc * C5);
  cast16_transpose_kernel<<<(Nc * Nc + 255) / 256, 256, 0, stream>>>(support0, StT, Nc, Nc);

  const long totA = (long)Bc * Nc * Hc * Lc;   // 317952 query rows

  // ---- self-attention block ----
  gemmNT(xh, wAh[0], bA[0], nullptr, qs, (int)Ms, Dc, Dc, Dc, Dc, Dc, 0, 0, 0, 0, 1, 0, 1);
  gemmNT(xh, wAh[1], bA[1], nullptr, ks, (int)Ms, Dc, Dc, Dc, Dc, Dc, 0, 0, 0, 0, 1, 0, 1);
  gemmNT(xh, wAh[2], bA[2], nullptr, vs, (int)Ms, Dc, Dc, Dc, Dc, Dc, 0, 0, 0, 0, 1, 0, 1);
  attn_kernel<<<(int)((totA + 255) / 256), 256, 0, stream>>>(qs, ks, vs, at1, Lc, Lc, Nc, Hc, totA);
  gemmNT(at1, wAh[3], bA[3], o32, nullptr, (int)Ms, Dc, Dc, Dc, Dc, Dc, 0, 0, 0, 0, 1, 0, 1);
  ln_kernel<<<(int)Ms, 128, 0, stream>>>(o32, x, g1, be1, x1f, x1h);

  // ---- cross-attention block ----
  gemmNT(x1h, wAh[4], bA[4], nullptr, qs, (int)Ms, Dc, Dc, Dc, Dc, Dc, 0, 0, 0, 0, 1, 0, 1);
  gemmNT(mh,  wAh[5], bA[5], nullptr, kc, (int)Mm, Dc, Dc, Dc, Dc, Dc, 0, 0, 0, 0, 1, 0, 1);
  gemmNT(mh,  wAh[6], bA[6], nullptr, vc, (int)Mm, Dc, Dc, Dc, Dc, Dc, 0, 0, 0, 0, 1, 0, 1);
  attn_kernel<<<(int)((totA + 255) / 256), 256, 0, stream>>>(qs, kc, vc, at1, Lc, LMc, Nc, Hc, totA);
  gemmNT(at1, wAh[7], bA[7], o32, nullptr, (int)Ms, Dc, Dc, Dc, Dc, Dc, 0, 0, 0, 0, 1, 0, 1);
  ln_kernel<<<(int)Ms, 128, 0, stream>>>(o32, x1f, g2, be2, x2f, x2h);

  // ---- FFN + out1 ----
  gemmNT(x2h, W1h, b1, nullptr, f1h, (int)Ms, DFFc, Dc, Dc, Dc, DFFc, 0, 0, 0, 0, 1, 1, 1);
  gemmNT(f1h, W2h, b2, y32, yh, (int)Ms, Dc, DFFc, DFFc, DFFc, Dc, 0, 0, 0, 0, 1, 0, 1);
  ln_kernel<<<(int)Ms, 128, 0, stream>>>(y32, x2f, g3, be3, out1, nullptr);

  // ---- dynamic adjacency As = y @ W3^T ----
  gemmNT(yh, W3h, b3, As32, nullptr, (int)Ms, Nc, Dc, Dc, Dc, Nc, 0, 0, 0, 0, 1, 0, 1);
  castN(As32, Ash, (long)NBt * Nc * Nc);

  // ---- GCN: build hT[b,l,n,5D] (transposed concat) ----
  {
    long rows = (long)NBt * Nc, n = rows * Dc;
    long bl = (n + 255) / 256; if (bl > 16384) bl = 16384;
    cast16_slice_kernel<<<(int)bl, 256, 0, stream>>>(data, hT, rows, Dc, C5);  // slice 0 = data
  }
  // h1a[n,d] = sum_{n'} support0[n',n] * data[n',d]  -> A = support0^T (shared), B NN-layout
  gemmSTR(StT, hT,        hT + 512,  Nc, Dc, Nc, Nc, 1, C5, 1, C5, 0,                 strideT, strideT, NBt);
  gemmSTR(StT, hT + 512,  hT + 1024, Nc, Dc, Nc, Nc, 1, C5, 1, C5, 0,                 strideT, strideT, NBt);
  // h1b[n,d] = sum_{n'} As[b,l,n',n] * data[n',d]   -> A = As^T via strides (per batch)
  gemmSTR(Ash, hT,        hT + 1536, Nc, Dc, Nc, 1, Nc, C5, 1, C5, (long)Nc * Nc,     strideT, strideT, NBt);
  gemmSTR(Ash, hT + 1536, hT + 2048, Nc, Dc, Nc, 1, Nc, C5, 1, C5, (long)Nc * Nc,     strideT, strideT, NBt);

  // ---- dts = hT @ Wg[l]^T + bg[l]  (per-(b,l) batch; weight index = batch % L) ----
  gemmNT(hT, Wgh, bg, outDts, nullptr, Nc, Dc, C5, C5, C5, Dc,
         strideT, (long)Dc * C5, (long)Nc * Dc, (long)Dc, Lc, 0, NBt);

  // ---- memory passthrough ----
  hipMemcpyAsync(outMem, memory_, (size_t)Mm * Dc * sizeof(float),
                 hipMemcpyDeviceToDevice, stream);
}
